// QuestionDescriminatorModel_35021163331986
// MI455X (gfx1250) — matrix operand order repaired and checked by reading, not compiled, not run on hardware
//
#include <hip/hip_runtime.h>
#include <hip/hip_bf16.h>
#include <math.h>

// ---------------- problem constants ----------------
constexpr int B_  = 2;
constexpr int S_  = 4096;
constexpr int D_  = 768;
constexpr int H_  = 12;
constexpr int DH_ = 64;
constexpr int L_  = 12;
constexpr int BSZ = 16;       // attention block size
constexpr int NB_ = S_ / BSZ; // 256
constexpr int NR_ = 2;
constexpr int DFF_ = 3072;
constexpr size_t NTOK = (size_t)B_ * S_;  // 8192

typedef __attribute__((ext_vector_type(16))) _Float16 v16h;
typedef __attribute__((ext_vector_type(8)))  float    v8f;

__device__ __forceinline__ float gelu_exact(float x) {
  return 0.5f * x * (1.0f + erff(x * 0.70710678118654752f));
}

// ---------------- f32 -> f16 convert ----------------
__global__ void cvt_f32_f16_kernel(const float* __restrict__ src,
                                   _Float16* __restrict__ dst, size_t n) {
  size_t i = (size_t)blockIdx.x * blockDim.x + threadIdx.x;
  if (i < n) dst[i] = (_Float16)src[i];
}

// ---------------- embedding + LN0 ----------------
__global__ __launch_bounds__(256) void embed_ln_kernel(
    const int* __restrict__ ids, const float* __restrict__ ew,
    const float* __restrict__ ep, const float* __restrict__ g,
    const float* __restrict__ bta, float* __restrict__ hout,
    _Float16* __restrict__ h16) {
  int row = blockIdx.x;              // 0..B*S-1
  int s = row % S_;
  int id = ids[row];
  int tid = threadIdx.x;
  __shared__ float red[256];
  float x[3];
  float lsum = 0.f;
#pragma unroll
  for (int i = 0; i < 3; ++i) {
    int d = tid + i * 256;
    x[i] = ew[(size_t)id * D_ + d] + ep[(size_t)s * D_ + d];
    lsum += x[i];
  }
  red[tid] = lsum; __syncthreads();
  for (int o = 128; o > 0; o >>= 1) { if (tid < o) red[tid] += red[tid + o]; __syncthreads(); }
  float mean = red[0] * (1.0f / D_);
  __syncthreads();
  float lv = 0.f;
#pragma unroll
  for (int i = 0; i < 3; ++i) { float t = x[i] - mean; lv += t * t; }
  red[tid] = lv; __syncthreads();
  for (int o = 128; o > 0; o >>= 1) { if (tid < o) red[tid] += red[tid + o]; __syncthreads(); }
  float rstd = rsqrtf(red[0] * (1.0f / D_) + 1e-12f);
#pragma unroll
  for (int i = 0; i < 3; ++i) {
    int d = tid + i * 256;
    float y = (x[i] - mean) * rstd * g[d] + bta[d];
    hout[(size_t)row * D_ + d] = y;
    h16[(size_t)row * D_ + d] = (_Float16)y;
  }
}

// ---------------- residual add + LN ----------------
__global__ __launch_bounds__(256) void add_ln_kernel(
    float* __restrict__ hbuf, const float* __restrict__ addv,
    const float* __restrict__ g, const float* __restrict__ bta,
    _Float16* __restrict__ h16) {
  int row = blockIdx.x;
  int tid = threadIdx.x;
  float* hp = hbuf + (size_t)row * D_;
  const float* ap = addv + (size_t)row * D_;
  __shared__ float red[256];
  float x[3];
  float lsum = 0.f;
#pragma unroll
  for (int i = 0; i < 3; ++i) { int d = tid + i * 256; x[i] = hp[d] + ap[d]; lsum += x[i]; }
  red[tid] = lsum; __syncthreads();
  for (int o = 128; o > 0; o >>= 1) { if (tid < o) red[tid] += red[tid + o]; __syncthreads(); }
  float mean = red[0] * (1.0f / D_);
  __syncthreads();
  float lv = 0.f;
#pragma unroll
  for (int i = 0; i < 3; ++i) { float t = x[i] - mean; lv += t * t; }
  red[tid] = lv; __syncthreads();
  for (int o = 128; o > 0; o >>= 1) { if (tid < o) red[tid] += red[tid + o]; __syncthreads(); }
  float rstd = rsqrtf(red[0] * (1.0f / D_) + 1e-12f);
#pragma unroll
  for (int i = 0; i < 3; ++i) {
    int d = tid + i * 256;
    float y = (x[i] - mean) * rstd * g[d] + bta[d];
    hp[d] = y;
    h16[(size_t)row * D_ + d] = (_Float16)y;
  }
}

// ---------------- WMMA GEMM: C[M,N] = A_f16[M,K] @ W_f16[K,N] + bias ----------------
// 128x128 C-tile / 256 threads (8 waves). Wave w: row tiles {2(w&3), 2(w&3)+1},
// col tiles {4(w>>2) .. 4(w>>2)+3}  =>  2 A-frags x 4 B-frags = 8 WMMA per K-step.
// fuse: 0 = none, 1 = exact GELU. scale applied after bias.
__global__ __launch_bounds__(256) void gemm_f16_wmma_kernel(
    const _Float16* __restrict__ A, const _Float16* __restrict__ W,
    const float* __restrict__ bias, float* __restrict__ Cf32,
    _Float16* __restrict__ Cf16, int M, int N, int K, float scale, int fuse) {
  int ntile = N >> 7;
  int tile = blockIdx.x;
  int m0 = (tile / ntile) << 7;
  int n0 = (tile % ntile) << 7;
  __shared__ __align__(16) _Float16 sA[128 * 32];      // [m][k] row-major
  __shared__ __align__(16) _Float16 sB[128 * 32];      // transposed: [n][k]
  int tid = threadIdx.x;
  int lane = tid & 31, wave = tid >> 5;
  int mr = (wave & 3) << 1;          // first of 2 row tiles
  int nc = (wave >> 2) << 2;         // first of 4 col tiles
  int l16 = lane & 15;
  int hi = lane >> 4;                // 0 or 1
  int akb2 = hi << 2;                // A u32 k-offset
  int bkb2 = hi << 2;                // B u32 k-offset
  const unsigned int* sAu = (const unsigned int*)sA;
  const unsigned int* sBu = (const unsigned int*)sB;

  // cooperative-load addressing (A: two uint4 per thread; W: two rows, paired store)
  int ar = tid >> 1;                  // A row 0..127
  int ac = (tid & 1) << 4;            // A col 0 or 16
  int wk2 = (tid & 15) << 1;          // W row pair base (even k)
  int wng = (tid >> 4) << 3;          // W col group base (8 cols)

  v8f acc[2][4];
#pragma unroll
  for (int ii = 0; ii < 2; ++ii)
#pragma unroll
    for (int jj = 0; jj < 4; ++jj) acc[ii][jj] = {};

  for (int k0 = 0; k0 < K; k0 += 32) {
    __syncthreads();
    { // A tile: 128x32 halves; each thread 16 contiguous halves (2 x uint4)
      const _Float16* src = A + (size_t)(m0 + ar) * K + k0 + ac;
      _Float16* dst = sA + ar * 32 + ac;
      *(uint4*)(dst)     = *(const uint4*)(src);
      *(uint4*)(dst + 8) = *(const uint4*)(src + 8);
      if (k0 + 32 < K) __builtin_prefetch(src + 32, 0, 3);
    }
    { // W tile: rows k (32) x cols n (128); store transposed [n][k] as packed pairs
      const _Float16* s0 = W + (size_t)(k0 + wk2) * N + n0 + wng;
      const _Float16* s1 = s0 + N;
      uint4 r0 = *(const uint4*)s0;
      uint4 r1 = *(const uint4*)s1;
      if (k0 + 32 < K) { __builtin_prefetch(s0 + (size_t)32 * N, 0, 3);
                         __builtin_prefetch(s1 + (size_t)32 * N, 0, 3); }
      const unsigned short* h0 = (const unsigned short*)&r0;
      const unsigned short* h1 = (const unsigned short*)&r1;
      unsigned int* bt = (unsigned int*)sB;
#pragma unroll
      for (int i = 0; i < 8; ++i)
        bt[(wng + i) * 16 + (wk2 >> 1)] =
            (unsigned int)h0[i] | ((unsigned int)h1[i] << 16);
    }
    __syncthreads();

    union Frag { v16h v; unsigned int u[8]; };
    Frag fa[2], fb[4];
#pragma unroll
    for (int ii = 0; ii < 2; ++ii) {
      int am = ((mr + ii) << 4) + l16;
#pragma unroll
      for (int j = 0; j < 4; ++j) {
        fa[ii].u[j]     = sAu[am * 16 + akb2 + j];
        fa[ii].u[j + 4] = sAu[am * 16 + akb2 + 8 + j];
      }
    }
#pragma unroll
    for (int jj = 0; jj < 4; ++jj) {
      int bn = ((nc + jj) << 4) + l16;
#pragma unroll
      for (int j = 0; j < 8; ++j) fb[jj].u[j] = sBu[bn * 16 + bkb2 + (j & 3) + ((j >> 2) << 3)];
    }
#pragma unroll
    for (int ii = 0; ii < 2; ++ii)
#pragma unroll
      for (int jj = 0; jj < 4; ++jj)
        acc[ii][jj] = __builtin_amdgcn_wmma_f32_16x16x32_f16(
            false, fa[ii].v, false, fb[jj].v, (short)0, acc[ii][jj], false, false);
  }

#pragma unroll
  for (int ii = 0; ii < 2; ++ii) {
#pragma unroll
    for (int jj = 0; jj < 4; ++jj) {
      int gn = n0 + ((nc + jj) << 4) + l16;
      float bn = bias[gn];
#pragma unroll
      for (int r = 0; r < 8; ++r) {
        int gm = m0 + ((mr + ii) << 4) + (hi << 3) + r;
        float x = (acc[ii][jj][r] + bn) * scale;
        if (fuse == 1) x = gelu_exact(x);
        size_t idx = (size_t)gm * N + gn;
        if (Cf32) Cf32[idx] = x;
        if (Cf16) Cf16[idx] = (_Float16)x;
      }
    }
  }
}

// ---------------- global-block attention: q rows of blocks 0 and NB-1 attend all keys --------
__global__ __launch_bounds__(256) void attn_global_kernel(
    const float* __restrict__ q, const float* __restrict__ k,
    const float* __restrict__ v, _Float16* __restrict__ o16) {
  int w = blockIdx.x;                 // B*H*2*16
  int r  = w & 15;
  int g  = (w >> 4) & 1;
  int hh = (w >> 5) % H_;
  int b  = w / (H_ * 2 * 16);
  int sq = g ? (S_ - 16 + r) : r;
  int tid = threadIdx.x;
  __shared__ float sc[S_];
  __shared__ float qs[DH_];
  __shared__ float red[256];
  __shared__ float oac[4][DH_];
  if (tid < DH_) qs[tid] = q[((size_t)b * S_ + sq) * D_ + hh * DH_ + tid];
  __syncthreads();
  float lmax = -3.4e38f;
  for (int j = tid; j < S_; j += 256) {
    const float* kp = k + ((size_t)b * S_ + j) * D_ + hh * DH_;
    float d = 0.f;
#pragma unroll 8
    for (int t = 0; t < DH_; ++t) d += qs[t] * kp[t];
    sc[j] = d;
    lmax = fmaxf(lmax, d);
  }
  red[tid] = lmax; __syncthreads();
  for (int o = 128; o > 0; o >>= 1) { if (tid < o) red[tid] = fmaxf(red[tid], red[tid + o]); __syncthreads(); }
  float mx = red[0];
  __syncthreads();
  float ls = 0.f;
  for (int j = tid; j < S_; j += 256) { float e = expf(sc[j] - mx); sc[j] = e; ls += e; }
  red[tid] = ls; __syncthreads();
  for (int o = 128; o > 0; o >>= 1) { if (tid < o) red[tid] += red[tid + o]; __syncthreads(); }
  float inv = 1.0f / red[0];
  int dq = tid & 63;
  int grp = tid >> 6;                 // 0..3
  float acc = 0.f;
  int j0 = grp * (S_ / 4), j1 = j0 + (S_ / 4);
  for (int j = j0; j < j1; ++j)
    acc += sc[j] * v[((size_t)b * S_ + j) * D_ + hh * DH_ + dq];
  oac[grp][dq] = acc;
  __syncthreads();
  if (tid < DH_) {
    float s = (oac[0][tid] + oac[1][tid] + oac[2][tid] + oac[3][tid]) * inv;
    o16[((size_t)b * S_ + sq) * D_ + hh * DH_ + tid] = (_Float16)s;
  }
}

// ---------------- mid-block attention: block i=n+1 attends 7 gathered blocks ----------------
__global__ __launch_bounds__(256) void attn_mid_kernel(
    const float* __restrict__ q, const float* __restrict__ k,
    const float* __restrict__ v, const int* __restrict__ rb,
    _Float16* __restrict__ o16) {
  int w = blockIdx.x;                 // B*H*(NB-2)
  int n  = w % (NB_ - 2);
  int hh = (w / (NB_ - 2)) % H_;
  int b  = w / ((NB_ - 2) * H_);
  int tid = threadIdx.x;
  __shared__ int kidx[7];
  __shared__ float qs[16 * DH_];
  __shared__ float kvs[7 * 16 * DH_];   // keys, then reused for values
  __shared__ float sc[16 * 7 * 16];     // 16 x 112 scores
  __shared__ float rinv[16];
  if (tid == 0) {
    kidx[0] = 0; kidx[1] = NB_ - 1;
    kidx[2] = n; kidx[3] = n + 1; kidx[4] = n + 2;
    kidx[5] = rb[n * NR_ + 0]; kidx[6] = rb[n * NR_ + 1];
  }
  __syncthreads();
  for (int t = tid; t < 16 * DH_; t += 256) {
    int r = t >> 6, d = t & 63;
    qs[t] = q[((size_t)b * S_ + (n + 1) * BSZ + r) * D_ + hh * DH_ + d];
  }
  for (int t = tid; t < 112 * DH_; t += 256) {
    int jj = t >> 6, d = t & 63;
    int sr = kidx[jj >> 4] * BSZ + (jj & 15);
    kvs[t] = k[((size_t)b * S_ + sr) * D_ + hh * DH_ + d];
  }
  __syncthreads();
  for (int e = tid; e < 16 * 112; e += 256) {
    int r = e / 112, j = e % 112;
    float d = 0.f;
#pragma unroll 8
    for (int t = 0; t < DH_; ++t) d += qs[r * DH_ + t] * kvs[j * DH_ + t];
    sc[e] = d;
  }
  __syncthreads();
  if (tid < 16) {
    float m = -3.4e38f;
    for (int j = 0; j < 112; ++j) m = fmaxf(m, sc[tid * 112 + j]);
    float s = 0.f;
    for (int j = 0; j < 112; ++j) { float e = expf(sc[tid * 112 + j] - m); sc[tid * 112 + j] = e; s += e; }
    rinv[tid] = 1.0f / s;
  }
  __syncthreads();
  for (int t = tid; t < 112 * DH_; t += 256) {  // reload same buffer with V
    int jj = t >> 6, d = t & 63;
    int sr = kidx[jj >> 4] * BSZ + (jj & 15);
    kvs[t] = v[((size_t)b * S_ + sr) * D_ + hh * DH_ + d];
  }
  __syncthreads();
  for (int e = tid; e < 16 * DH_; e += 256) {
    int r = e >> 6, d = e & 63;
    float a = 0.f;
#pragma unroll 8
    for (int j = 0; j < 112; ++j) a += sc[r * 112 + j] * kvs[j * DH_ + d];
    o16[((size_t)b * S_ + (n + 1) * BSZ + r) * D_ + hh * DH_ + d] = (_Float16)(a * rinv[r]);
  }
}

// ---------------- start / end logits ----------------
__global__ __launch_bounds__(256) void logits_kernel(
    const float* __restrict__ h, const float* __restrict__ sw,
    const float* __restrict__ ew, float* __restrict__ out) {
  int row = blockIdx.x * 256 + threadIdx.x;
  if (row >= (int)NTOK) return;
  const float* hp = h + (size_t)row * D_;
  float a = 0.f, c = 0.f;
  for (int d = 0; d < D_; ++d) { float x = hp[d]; a += x * sw[d]; c += x * ew[d]; }
  out[row] = a;
  out[NTOK + row] = c;
}

// ---------------- pooled tanh + 2-layer discriminator head ----------------
__global__ __launch_bounds__(256) void disc_head_kernel(
    const float* __restrict__ h, const float* __restrict__ Wp,
    const float* __restrict__ bp, const float* __restrict__ Wd1,
    const float* __restrict__ bd1, const float* __restrict__ Wd2,
    const float* __restrict__ bd2, float* __restrict__ out) {
  __shared__ float pooled[2][D_];
  __shared__ float d1[2][20];
  int tid = threadIdx.x;
  for (int e = tid; e < 2 * D_; e += 256) {
    int b = e / D_, c = e % D_;
    const float* hp = h + (size_t)b * S_ * D_;  // token 0
    float a = bp[c];
    for (int d = 0; d < D_; ++d) a += hp[d] * Wp[(size_t)d * D_ + c];
    pooled[b][c] = tanhf(a);
  }
  __syncthreads();
  for (int e = tid; e < 2 * 20; e += 256) {
    int b = e / 20, c = e % 20;
    float a = bd1[c];
    for (int d = 0; d < D_; ++d) a += pooled[b][d] * Wd1[d * 20 + c];
    d1[b][c] = fmaxf(a, 0.f);
  }
  __syncthreads();
  if (tid < 4) {
    int b = tid >> 1, c = tid & 1;
    float a = bd2[c];
    for (int d = 0; d < 20; ++d) a += d1[b][d] * Wd2[d * 2 + c];
    out[b * 2 + c] = a;
  }
}

// ---------------- host orchestration ----------------
extern "C" void kernel_launch(void* const* d_in, const int* in_sizes, int n_in,
                              void* d_out, int out_size, void* d_ws, size_t ws_size,
                              hipStream_t stream) {
  const int*   input_ids = (const int*)d_in[0];
  const int*   rand_blk  = (const int*)d_in[1];
  const float* emb_word  = (const float*)d_in[2];
  const float* emb_pos   = (const float*)d_in[3];
  const float* ln0_s     = (const float*)d_in[4];
  const float* ln0_b     = (const float*)d_in[5];
  const float* Wq = (const float*)d_in[6];  const float* bq = (const float*)d_in[7];
  const float* Wk = (const float*)d_in[8];  const float* bk = (const float*)d_in[9];
  const float* Wv = (const float*)d_in[10]; const float* bv = (const float*)d_in[11];
  const float* Wo = (const float*)d_in[12]; const float* bo = (const float*)d_in[13];
  const float* ln1_s = (const float*)d_in[14]; const float* ln1_b = (const float*)d_in[15];
  const float* W1 = (const float*)d_in[16]; const float* b1 = (const float*)d_in[17];
  const float* W2 = (const float*)d_in[18]; const float* b2 = (const float*)d_in[19];
  const float* ln2_s = (const float*)d_in[20]; const float* ln2_b = (const float*)d_in[21];
  const float* Wp = (const float*)d_in[22]; const float* bp = (const float*)d_in[23];
  const float* start_w = (const float*)d_in[24];
  const float* end_w   = (const float*)d_in[25];
  const float* Wd1 = (const float*)d_in[26]; const float* bd1 = (const float*)d_in[27];
  const float* Wd2 = (const float*)d_in[28]; const float* bd2 = (const float*)d_in[29];
  float* out = (float*)d_out;

  // workspace layout (~206 MB)
  char* ws = (char*)d_ws;
  constexpr size_t SZ_F32 = NTOK * D_ * sizeof(float);       // 25,165,824
  constexpr size_t SZ_F16 = NTOK * D_ * sizeof(_Float16);    // 12,582,912
  constexpr size_t SZ_MID = NTOK * DFF_ * sizeof(_Float16);  // 50,331,648
  constexpr size_t SZ_WGT = (size_t)D_ * DFF_ * sizeof(_Float16); // 4,718,592
  size_t off = 0;
  float*    hf32   = (float*)(ws + off); off += SZ_F32;
  _Float16* hf16   = (_Float16*)(ws + off); off += SZ_F16;
  float*    qf32   = (float*)(ws + off); off += SZ_F32;
  float*    kf32   = (float*)(ws + off); off += SZ_F32;
  float*    vf32   = (float*)(ws + off); off += SZ_F32;
  float*    tmpf32 = (float*)(ws + off); off += SZ_F32;
  _Float16* attn16 = (_Float16*)(ws + off); off += SZ_F16;
  _Float16* mid16  = (_Float16*)(ws + off); off += SZ_MID;
  _Float16* wgt16  = (_Float16*)(ws + off); off += SZ_WGT;
  (void)off; (void)ws_size; (void)in_sizes; (void)n_in; (void)out_size;

  auto cvt = [&](const float* s, _Float16* d, size_t n) {
    cvt_f32_f16_kernel<<<(unsigned)((n + 255) / 256), 256, 0, stream>>>(s, d, n);
  };
  auto gemm = [&](const _Float16* A, const _Float16* W, const float* bias,
                  float* Cf32, _Float16* Cf16, int M, int N, int K,
                  float scale, int fuse) {
    gemm_f16_wmma_kernel<<<(unsigned)((M / 128) * (N / 128)), 256, 0, stream>>>(
        A, W, bias, Cf32, Cf16, M, N, K, scale, fuse);
  };

  // embedding + LN0
  embed_ln_kernel<<<(unsigned)NTOK, 256, 0, stream>>>(
      input_ids, emb_word, emb_pos, ln0_s, ln0_b, hf32, hf16);

  const float qscale = 0.125f;  // 1/sqrt(DH)
  for (int l = 0; l < L_; ++l) {
    const float* wq = Wq + (size_t)l * D_ * D_;   const float* bq_ = bq + (size_t)l * D_;
    const float* wk = Wk + (size_t)l * D_ * D_;   const float* bk_ = bk + (size_t)l * D_;
    const float* wv = Wv + (size_t)l * D_ * D_;   const float* bv_ = bv + (size_t)l * D_;
    const float* wo = Wo + (size_t)l * D_ * D_;   const float* bo_ = bo + (size_t)l * D_;
    const float* w1 = W1 + (size_t)l * D_ * DFF_; const float* b1_ = b1 + (size_t)l * DFF_;
    const float* w2 = W2 + (size_t)l * DFF_ * D_; const float* b2_ = b2 + (size_t)l * D_;
    const float* l1s = ln1_s + (size_t)l * D_; const float* l1b = ln1_b + (size_t)l * D_;
    const float* l2s = ln2_s + (size_t)l * D_; const float* l2b = ln2_b + (size_t)l * D_;

    // Q/K/V projections (Q fused with 1/sqrt(DH) scale)
    cvt(wq, wgt16, (size_t)D_ * D_);
    gemm(hf16, wgt16, bq_, qf32, nullptr, (int)NTOK, D_, D_, qscale, 0);
    cvt(wk, wgt16, (size_t)D_ * D_);
    gemm(hf16, wgt16, bk_, kf32, nullptr, (int)NTOK, D_, D_, 1.0f, 0);
    cvt(wv, wgt16, (size_t)D_ * D_);
    gemm(hf16, wgt16, bv_, vf32, nullptr, (int)NTOK, D_, D_, 1.0f, 0);

    // block-sparse attention -> attn16 (f16 context, feeds Wo GEMM)
    attn_global_kernel<<<B_ * H_ * 2 * 16, 256, 0, stream>>>(qf32, kf32, vf32, attn16);
    attn_mid_kernel<<<B_ * H_ * (NB_ - 2), 256, 0, stream>>>(qf32, kf32, vf32, rand_blk, attn16);

    // output projection + residual LN
    cvt(wo, wgt16, (size_t)D_ * D_);
    gemm(attn16, wgt16, bo_, tmpf32, nullptr, (int)NTOK, D_, D_, 1.0f, 0);
    add_ln_kernel<<<(unsigned)NTOK, 256, 0, stream>>>(hf32, tmpf32, l1s, l1b, hf16);

    // FFN: GELU fused in GEMM1 (f16 out only), GEMM2 back to D, residual LN
    cvt(w1, wgt16, (size_t)D_ * DFF_);
    gemm(hf16, wgt16, b1_, nullptr, mid16, (int)NTOK, DFF_, D_, 1.0f, 1);
    cvt(w2, wgt16, (size_t)DFF_ * D_);
    gemm(mid16, wgt16, b2_, tmpf32, nullptr, (int)NTOK, D_, DFF_, 1.0f, 0);
    add_ln_kernel<<<(unsigned)NTOK, 256, 0, stream>>>(hf32, tmpf32, l2s, l2b, hf16);
  }

  // heads
  logits_kernel<<<(unsigned)((NTOK + 255) / 256), 256, 0, stream>>>(hf32, start_w, end_w, out);
  disc_head_kernel<<<1, 256, 0, stream>>>(hf32, Wp, bp, Wd1, bd1, Wd2, bd2,
                                          out + 2 * NTOK);
}
